// GAT_FP_20675972563340
// MI455X (gfx1250) — matrix operand order repaired and checked by reading, not compiled, not run on hardware
//
#include <hip/hip_runtime.h>
#include <math.h>
#include <stdint.h>

#define NN   12000
#define EE   192000
#define TT   120
#define DIN  192
#define NDLG (NN / TT)   // 100

typedef __attribute__((ext_vector_type(16))) __bf16 v16bf;
typedef __attribute__((ext_vector_type(8)))  float  v8f;

// ---------- helpers ----------
__device__ __forceinline__ unsigned short bfbits(float f) {
  unsigned u = __float_as_uint(f);
  return (unsigned short)((u + 0x7FFFu + ((u >> 16) & 1u)) >> 16);   // RNE
}
__device__ __forceinline__ __bf16 f2bf(float f) {
  unsigned short s = bfbits(f);
  return __builtin_bit_cast(__bf16, s);
}
__device__ __forceinline__ uint2 pack4bf(float4 v) {
  uint2 r;
  r.x = (unsigned)bfbits(v.x) | ((unsigned)bfbits(v.y) << 16);
  r.y = (unsigned)bfbits(v.z) | ((unsigned)bfbits(v.w) << 16);
  return r;
}

struct BFrag { uint4 lo, hi; };

// A fragment (16-bit A 16x32 ISA layout): lane row = lane&15, kb=(lane>>4)*8;
// elements 0..7 at p[0..7], elements 8..15 at p[16..23]
__device__ __forceinline__ v16bf load_fragA(const __bf16* p) {
  BFrag t;
  t.lo = *(const uint4*)(p);
  t.hi = *(const uint4*)(p + 16);
  return __builtin_bit_cast(v16bf, t);
}
// B fragment: lane col = lane&15, kb=(lane>>4)*8; elements 0..7 at p[0..7], 8..15 at p[16..23] (N-major LDS)
__device__ __forceinline__ v16bf load_fragB(const __bf16* p) {
  BFrag t;
  t.lo = *(const uint4*)(p);
  t.hi = *(const uint4*)(p + 16);
  return __builtin_bit_cast(v16bf, t);
}

__device__ __forceinline__ void atomicMaxFloat(float* addr, float v) {
  if (v >= 0.f) atomicMax((int*)addr, __float_as_int(v));
  else          atomicMin((unsigned int*)addr, __float_as_uint(v));
}

// ---------- WMMA GEMM: C[m,n] = (sum_k A[m,k]*B[k,n] + bias[n]) * rowscale[m] ----------
// A: f32 [M x K] (bf16-converted while staging); Bt: bf16 [Nout x K] (pre-transposed, N-major).
// grid = (M/32, Nout/64), block = 128 (4 waves). Each wave: two 16x16 tiles (rows 0-15 / 16-31),
// two WMMAs per 32-wide K step sharing one B fragment.
__global__ __launch_bounds__(128) void k_gemm(
    const float* __restrict__ A, const __bf16* __restrict__ Bt,
    const float* __restrict__ bias, const float* __restrict__ rowscale,
    float* __restrict__ C, int K, int Nout, int ldc) {
  __shared__ __attribute__((aligned(16))) __bf16 ldsA[32 * 32];
  __shared__ __attribute__((aligned(16))) __bf16 ldsB[64 * 32];   // [n][k]
  __shared__ __attribute__((aligned(16))) float  ldsC[32 * 64];

  const int tid  = threadIdx.x;
  const int wave = tid >> 5;
  const int lane = tid & 31;
  const int mrow0 = blockIdx.x * 32;
  const int ncol0 = blockIdx.y * 64;

  const int nloc = lane & 15;
  const int kbA  = (lane >> 4) * 8;
  const int kbB  = (lane >> 4) * 8;
  const int ncl  = wave * 16 + nloc;        // local column 0..63
  const int mbase = (lane >> 4) * 8;

  v8f acc0 = {}, acc1 = {};
  const float bv = bias ? bias[ncol0 + ncl] : 0.f;
#pragma unroll
  for (int r = 0; r < 8; ++r) { acc0[r] = bv; acc1[r] = bv; }

  for (int k0 = 0; k0 < K; k0 += 32) {
    // stage A 32x32 f32 -> bf16: 256 float4 chunks, 2 per thread, ds_store_b64 each
#pragma unroll
    for (int i = 0; i < 2; ++i) {
      int cid = tid + i * 128;              // 0..255
      int r = cid >> 3, c4 = (cid & 7) * 4;
      float4 v = *(const float4*)(A + (size_t)(mrow0 + r) * K + k0 + c4);
      *(uint2*)(ldsA + r * 32 + c4) = pack4bf(v);
    }
    // stage B 64x32 bf16 (already N-major in global): 256 uint4 chunks, 2 per thread
#pragma unroll
    for (int i = 0; i < 2; ++i) {
      int cid = tid + i * 128;              // 0..255
      int n = cid >> 2, k8 = (cid & 3) * 8;
      *(uint4*)(ldsB + n * 32 + k8) =
          *(const uint4*)(Bt + (size_t)(ncol0 + n) * K + k0 + k8);
    }
    __syncthreads();
    v16bf bfr = load_fragB(&ldsB[(wave * 16 + nloc) * 32 + kbB]);
    v16bf a0  = load_fragA(&ldsA[nloc * 32 + kbA]);
    v16bf a1  = load_fragA(&ldsA[(16 + nloc) * 32 + kbA]);
    acc0 = __builtin_amdgcn_wmma_f32_16x16x32_bf16(false, a0, false, bfr,
                                                   (short)0, acc0, false, false);
    acc1 = __builtin_amdgcn_wmma_f32_16x16x32_bf16(false, a1, false, bfr,
                                                   (short)0, acc1, false, false);
    __syncthreads();
  }

  // spill accumulators to LDS (C tile 32x64), then coalesced float4 stores
#pragma unroll
  for (int r = 0; r < 8; ++r) {
    ldsC[(mbase + r) * 64 + ncl]      = acc0[r];
    ldsC[(16 + mbase + r) * 64 + ncl] = acc1[r];
  }
  __syncthreads();
  const int row = tid >> 2;                 // 0..31
  const int cb  = (tid & 3) * 16;           // 0,16,32,48
  const float rs = rowscale ? rowscale[mrow0 + row] : 1.f;
  float* cp = C + (size_t)(mrow0 + row) * ldc + ncol0 + cb;
#pragma unroll
  for (int j = 0; j < 4; ++j) {
    float4 v = *(const float4*)(ldsC + row * 64 + cb + j * 4);
    v.x *= rs; v.y *= rs; v.z *= rs; v.w *= rs;
    *(float4*)(cp + j * 4) = v;
  }
}

// ---------- small utility kernels ----------
// convert f32 weight [K x N] -> bf16 transposed [N x K]
__global__ void k_f2bf_t(const float* __restrict__ in, __bf16* __restrict__ out,
                         int K, int N) {
  int i = blockIdx.x * blockDim.x + threadIdx.x;
  if (i >= K * N) return;
  int n = i / K, k = i % K;
  out[i] = f2bf(in[(size_t)k * N + n]);
}
__global__ void k_fill(float* __restrict__ p, float v, int n) {
  int i = blockIdx.x * blockDim.x + threadIdx.x;
  if (i < n) p[i] = v;
}
__global__ void k_relu(float* __restrict__ p, int n) {
  int i = blockIdx.x * blockDim.x + threadIdx.x;
  if (i < n) p[i] = fmaxf(p[i], 0.f);
}
__global__ void k_degree(const int* __restrict__ src, const int* __restrict__ dst,
                         float* __restrict__ dO, float* __restrict__ dI) {
  int e = blockIdx.x * blockDim.x + threadIdx.x;
  if (e < EE) { atomicAdd(&dO[src[e]], 1.f); atomicAdd(&dI[dst[e]], 1.f); }
}
__global__ void k_rsqrt_clamp(float* __restrict__ d, int n) {
  int i = blockIdx.x * blockDim.x + threadIdx.x;
  if (i < n) d[i] = rsqrtf(fmaxf(d[i], 1.f));
}

// ---------- BiLSTM: one wave per (dialogue, direction); 2 lanes per gate ----------
__global__ void k_lstm(const float* __restrict__ x,
                       const float* __restrict__ Wih_f, const float* __restrict__ Whh_f,
                       const float* __restrict__ bih_f, const float* __restrict__ bhh_f,
                       const float* __restrict__ Wih_b, const float* __restrict__ Whh_b,
                       const float* __restrict__ bih_b, const float* __restrict__ bhh_b,
                       float* __restrict__ nf) {
  __shared__ float sW[2][16 * DIN];
  for (int i = threadIdx.x; i < 16 * DIN; i += blockDim.x) {
    sW[0][i] = Wih_f[i]; sW[1][i] = Wih_b[i];
  }
  __syncthreads();
  int gw = blockIdx.x * (blockDim.x >> 5) + (threadIdx.x >> 5);
  int lane = threadIdx.x & 31;
  int dlg = gw >> 1, dir = gw & 1;
  if (dlg >= NDLG) return;
  const float* Whh = dir ? Whh_b : Whh_f;
  const float* bih = dir ? bih_b : bih_f;
  const float* bhh = dir ? bhh_b : bhh_f;
  const float* Wih = sW[dir];
  const int gate = lane >> 1, half = lane & 1;
  const float bsum = bih[gate] + bhh[gate];
  float h[4] = {0, 0, 0, 0};
  float c = 0.f;
  const int j = lane & 3;
  for (int t = 0; t < TT; ++t) {
    int tt = dir ? (TT - 1 - t) : t;
    const float* xr = x + (size_t)(dlg * TT + tt) * DIN;
    float z = 0.f;
    const float* wr = Wih + gate * DIN + half * 96;
    const float* xp = xr + half * 96;
#pragma unroll 8
    for (int i = 0; i < 96; ++i) z += wr[i] * xp[i];
    z += __shfl_xor(z, 1);
#pragma unroll
    for (int q = 0; q < 4; ++q) z += Whh[gate * 4 + q] * h[q];
    z += bsum;
    float zi = __shfl(z, 2 * j);
    float zf = __shfl(z, 2 * (4 + j));
    float zg = __shfl(z, 2 * (8 + j));
    float zo = __shfl(z, 2 * (12 + j));
    float ii = 1.f / (1.f + __expf(-zi));
    float ff = 1.f / (1.f + __expf(-zf));
    float gg = tanhf(zg);
    float oo = 1.f / (1.f + __expf(-zo));
    c = ff * c + ii * gg;
    float hn = oo * tanhf(c);
#pragma unroll
    for (int q = 0; q < 4; ++q) h[q] = __shfl(hn, q);
    if (lane < 4) nf[(size_t)(dlg * TT + tt) * 8 + dir * 4 + lane] = hn;
  }
}

// ---------- GraphConv edge scatter: h1[dst] += hw[src] ----------
__global__ void k_scatter_gc(const float* __restrict__ hw, const int* __restrict__ src,
                             const int* __restrict__ dst, float* __restrict__ h1) {
  int i = blockIdx.x * blockDim.x + threadIdx.x;
  const int per = DIN / 4;
  int e = i / per;
  if (e >= EE) return;
  int c = (i % per) * 4;
  const float4 v = *(const float4*)(hw + (size_t)src[e] * DIN + c);
  float* o = h1 + (size_t)dst[e] * DIN + c;
  atomicAdd(o + 0, v.x); atomicAdd(o + 1, v.y);
  atomicAdd(o + 2, v.z); atomicAdd(o + 3, v.w);
}

// ---------- GraphConv finalize + row L1 normalize (block = 192 = one node) ----------
__global__ void k_gc_finalize(const float* __restrict__ sFT, const float* __restrict__ h1,
                              const float* __restrict__ rsid, const float* __restrict__ bgc,
                              float* __restrict__ h) {
  int n = blockIdx.x, c = threadIdx.x;
  float val = 0.5f * (sFT[(size_t)n * DIN + c] + h1[(size_t)n * DIN + c] * rsid[n] + bgc[c]);
  float a = fabsf(val);
#pragma unroll
  for (int o = 16; o > 0; o >>= 1) a += __shfl_xor(a, o);
  __shared__ float wsum[6];
  if ((c & 31) == 0) wsum[c >> 5] = a;
  __syncthreads();
  float tot = wsum[0] + wsum[1] + wsum[2] + wsum[3] + wsum[4] + wsum[5];
  h[(size_t)n * DIN + c] = val / fmaxf(tot, 1e-12f);
}

// ---------- GATv2 edge kernels (H = 16 heads) ----------
__global__ void k_gat_logits(const float* __restrict__ fs, const float* __restrict__ fd,
                             const float* __restrict__ attn, const int* __restrict__ src,
                             const int* __restrict__ dst, float* __restrict__ lg,
                             float* __restrict__ mb, int F) {
  int i = blockIdx.x * blockDim.x + threadIdx.x;
  if (i >= EE * 16) return;
  int e = i >> 4, hh = i & 15;
  int s = src[e], d = dst[e];
  const float* ps = fs + ((size_t)s * 16 + hh) * F;
  const float* pd = fd + ((size_t)d * 16 + hh) * F;
  const float* pa = attn + hh * F;
  float L = 0.f;
#pragma unroll 4
  for (int f = 0; f < F; ++f) {
    float v = ps[f] + pd[f];
    v = (v > 0.f) ? v : 0.2f * v;
    L += pa[f] * v;
  }
  lg[i] = L;
  atomicMaxFloat(&mb[d * 16 + hh], L);
}

__global__ void k_gat_exps(const int* __restrict__ dst, const float* __restrict__ mb,
                           float* __restrict__ lg, float* __restrict__ sb) {
  int i = blockIdx.x * blockDim.x + threadIdx.x;
  if (i >= EE * 16) return;
  int e = i >> 4, hh = i & 15;
  int d = dst[e];
  float ex = __expf(lg[i] - mb[d * 16 + hh]);
  lg[i] = ex;
  atomicAdd(&sb[d * 16 + hh], ex);
}

__global__ void k_gat_scatter(const float* __restrict__ fs, const float* __restrict__ ex,
                              const float* __restrict__ sb, const int* __restrict__ src,
                              const int* __restrict__ dst, float* __restrict__ out, int F) {
  int i = blockIdx.x * blockDim.x + threadIdx.x;
  const int per = F >> 2;
  int eh = i / per;
  if (eh >= EE * 16) return;
  int c4 = (i % per) * 4;
  int e = eh >> 4, hh = eh & 15;
  int s = src[e], d = dst[e];
  float alpha = ex[eh] / sb[d * 16 + hh];
  const float4 v = *(const float4*)(fs + ((size_t)s * 16 + hh) * F + c4);
  float* o = out + ((size_t)d * 16 + hh) * F + c4;
  atomicAdd(o + 0, alpha * v.x); atomicAdd(o + 1, alpha * v.y);
  atomicAdd(o + 2, alpha * v.z); atomicAdd(o + 3, alpha * v.w);
}

// ---------- classifier: concat[hg(64) | nf(8) | h3(64)] @ W_lin(136x6) + b ----------
__global__ void k_classifier(const float* __restrict__ hg, const float* __restrict__ nf,
                             const float* __restrict__ h3, const float* __restrict__ Wl,
                             const float* __restrict__ bl, float* __restrict__ out) {
  int i = blockIdx.x * blockDim.x + threadIdx.x;
  if (i >= NN * 6) return;
  int n = i / 6, o = i % 6;
  float acc = bl[o];
#pragma unroll 8
  for (int k = 0; k < 64; ++k) acc += hg[(size_t)n * 64 + k] * Wl[k * 6 + o];
#pragma unroll
  for (int k = 0; k < 8; ++k)  acc += nf[(size_t)n * 8 + k] * Wl[(64 + k) * 6 + o];
#pragma unroll 8
  for (int k = 0; k < 64; ++k) acc += h3[(size_t)n * 64 + k] * Wl[(72 + k) * 6 + o];
  out[i] = acc;
}

// ---------- host orchestration ----------
extern "C" void kernel_launch(void* const* d_in, const int* in_sizes, int n_in,
                              void* d_out, int out_size, void* d_ws, size_t ws_size,
                              hipStream_t stream) {
  (void)in_sizes; (void)n_in; (void)out_size; (void)ws_size;
  const float* text   = (const float*)d_in[0];
  const float* audio  = (const float*)d_in[1];
  const float* vision = (const float*)d_in[2];
  const int*   esrc   = (const int*)d_in[3];
  const int*   edst   = (const int*)d_in[4];
  const float* W_audio=(const float*)d_in[5];  const float* b_audio=(const float*)d_in[6];
  const float* W_vision=(const float*)d_in[7]; const float* b_vision=(const float*)d_in[8];
  const float* W_text=(const float*)d_in[9];   const float* b_text=(const float*)d_in[10];
  const float* Wih_f=(const float*)d_in[11];   const float* Whh_f=(const float*)d_in[12];
  const float* bih_f=(const float*)d_in[13];   const float* bhh_f=(const float*)d_in[14];
  const float* Wih_b=(const float*)d_in[15];   const float* Whh_b=(const float*)d_in[16];
  const float* bih_b=(const float*)d_in[17];   const float* bhh_b=(const float*)d_in[18];
  const float* W_gc=(const float*)d_in[19];    const float* b_gc=(const float*)d_in[20];
  const float* Wsrc2=(const float*)d_in[21];   const float* bsrc2=(const float*)d_in[22];
  const float* Wdst2=(const float*)d_in[23];   const float* bdst2=(const float*)d_in[24];
  const float* attn2=(const float*)d_in[25];   const float* Wres2=(const float*)d_in[26];
  const float* Wsrc0=(const float*)d_in[27];   const float* bsrc0=(const float*)d_in[28];
  const float* Wdst0=(const float*)d_in[29];   const float* bdst0=(const float*)d_in[30];
  const float* attn0=(const float*)d_in[31];   const float* Wres0=(const float*)d_in[32];
  const float* Wsrc1=(const float*)d_in[33];   const float* bsrc1=(const float*)d_in[34];
  const float* Wdst1=(const float*)d_in[35];   const float* bdst1=(const float*)d_in[36];
  const float* attn1=(const float*)d_in[37];   const float* Wres1=(const float*)d_in[38];
  const float* W_lin=(const float*)d_in[39];   const float* b_lin=(const float*)d_in[40];

  char* ws = (char*)d_ws;
  size_t off = 0;
  auto alloc = [&](size_t bytes) -> char* {
    off = (off + 255) & ~(size_t)255;
    char* p = ws + off; off += bytes; return p;
  };
  auto allocf = [&](size_t n) { return (float*)alloc(n * 4); };
  auto allocb = [&](size_t n) { return (__bf16*)alloc(n * 2); };
  // convert + transpose weight [K x N] -> bf16 [N x K]
  auto toBfT = [&](const float* srcp, int K, int N) -> __bf16* {
    __bf16* p = allocb((size_t)K * N);
    k_f2bf_t<<<(K * N + 255) / 256, 256, 0, stream>>>(srcp, p, K, N);
    return p;
  };

  __bf16* wt_text  = toBfT(W_text,  1024, 64);
  __bf16* wt_audio = toBfT(W_audio,  512, 64);
  __bf16* wt_vis   = toBfT(W_vision,1024, 64);
  __bf16* wt_gc    = toBfT(W_gc,     192, 192);
  __bf16* wt_s2 = toBfT(Wsrc2, 192, 64); __bf16* wt_d2 = toBfT(Wdst2, 192, 64);
  __bf16* wt_r2 = toBfT(Wres2, 192, 64);
  __bf16* wt_s0 = toBfT(Wsrc0, 192, 512); __bf16* wt_d0 = toBfT(Wdst0, 192, 512);
  __bf16* wt_r0 = toBfT(Wres0, 192, 512);
  __bf16* wt_s1 = toBfT(Wsrc1, 512, 64); __bf16* wt_d1 = toBfT(Wdst1, 512, 64);
  __bf16* wt_r1 = toBfT(Wres1, 512, 64);

  float* stackFT = allocf((size_t)NN * DIN);
  float* newFeat = allocf((size_t)NN * 8);
  float* degO = allocf(NN);
  float* degI = allocf(NN);
  float* hw   = allocf((size_t)NN * DIN);
  float* h1   = allocf((size_t)NN * DIN);
  float* hN   = allocf((size_t)NN * DIN);
  float* fsB  = allocf((size_t)NN * 512);
  float* fdB  = allocf((size_t)NN * 512);
  float* h0   = allocf((size_t)NN * 512);
  float* exB  = allocf((size_t)EE * 16);
  float* mB   = allocf((size_t)NN * 16);
  float* sB   = allocf((size_t)NN * 16);
  float* h3   = allocf((size_t)NN * 64);
  float* hg   = allocf((size_t)NN * 64);

  auto gemm = [&](const float* A, const __bf16* Bt, const float* bias, const float* rs,
                  float* C, int K, int Nout, int ldc) {
    dim3 g(NN / 32, Nout / 64);
    k_gemm<<<g, 128, 0, stream>>>(A, Bt, bias, rs, C, K, Nout, ldc);
  };

  // 1) modality encoders -> stackFT = [t | a | v]
  gemm(text,   wt_text,  b_text,   nullptr, stackFT + 0,   1024, 64, DIN);
  gemm(audio,  wt_audio, b_audio,  nullptr, stackFT + 64,   512, 64, DIN);
  gemm(vision, wt_vis,   b_vision, nullptr, stackFT + 128, 1024, 64, DIN);

  // 2) BiLSTM -> newFeat [N, 8]
  k_lstm<<<(NDLG * 2 + 7) / 8, 256, 0, stream>>>(stackFT, Wih_f, Whh_f, bih_f, bhh_f,
                                                 Wih_b, Whh_b, bih_b, bhh_b, newFeat);

  // 3) degrees -> rsqrt(max(deg,1))
  k_fill<<<(NN + 255) / 256, 256, 0, stream>>>(degO, 0.f, NN);
  k_fill<<<(NN + 255) / 256, 256, 0, stream>>>(degI, 0.f, NN);
  k_degree<<<(EE + 255) / 256, 256, 0, stream>>>(esrc, edst, degO, degI);
  k_rsqrt_clamp<<<(NN + 255) / 256, 256, 0, stream>>>(degO, NN);
  k_rsqrt_clamp<<<(NN + 255) / 256, 256, 0, stream>>>(degI, NN);

  // 4) GraphConv (rowscale fused in GEMM epilogue)
  gemm(stackFT, wt_gc, nullptr, degO, hw, DIN, DIN, DIN);
  k_fill<<<(NN * DIN + 255) / 256, 256, 0, stream>>>(h1, 0.f, NN * DIN);
  k_scatter_gc<<<(EE * 48 + 255) / 256, 256, 0, stream>>>(hw, esrc, edst, h1);
  k_gc_finalize<<<NN, DIN, 0, stream>>>(stackFT, h1, degI, b_gc, hN);

  // 5) GATv2 layers
  auto gat = [&](const float* hin, int K, int F, const __bf16* Ws, const float* bs,
                 const __bf16* Wd, const float* bd, const float* attn,
                 const __bf16* Wr, float* outp) {
    int HF = 16 * F;
    gemm(hin, Ws, bs, nullptr, fsB, K, HF, HF);
    gemm(hin, Wd, bd, nullptr, fdB, K, HF, HF);
    gemm(hin, Wr, nullptr, nullptr, outp, K, HF, HF);   // residual init of accumulator
    k_fill<<<(NN * 16 + 255) / 256, 256, 0, stream>>>(mB, -INFINITY, NN * 16);
    k_fill<<<(NN * 16 + 255) / 256, 256, 0, stream>>>(sB, 0.f, NN * 16);
    int nL = EE * 16;
    k_gat_logits<<<(nL + 255) / 256, 256, 0, stream>>>(fsB, fdB, attn, esrc, edst, exB, mB, F);
    k_gat_exps<<<(nL + 255) / 256, 256, 0, stream>>>(edst, mB, exB, sB);
    int nS = EE * 16 * (F / 4);
    k_gat_scatter<<<(nS + 255) / 256, 256, 0, stream>>>(fsB, exB, sB, esrc, edst, outp, F);
    k_relu<<<(NN * HF + 255) / 256, 256, 0, stream>>>(outp, NN * HF);
  };

  gat(hN, DIN, 4,  wt_s2, bsrc2, wt_d2, bdst2, attn2, wt_r2, h3);   // cross-modal branch
  gat(hN, DIN, 32, wt_s0, bsrc0, wt_d0, bdst0, attn0, wt_r0, h0);   // layer 0: 192 -> 512
  gat(h0, 512, 4,  wt_s1, bsrc1, wt_d1, bdst1, attn1, wt_r1, hg);   // layer 1: 512 -> 64

  // 6) classifier
  k_classifier<<<(NN * 6 + 255) / 256, 256, 0, stream>>>(hg, newFeat, h3, W_lin, b_lin,
                                                         (float*)d_out);
}
